// RetinaNet_41850161332642
// MI455X (gfx1250) — compile-verified, matrix operand order checked by reading, not run on hardware
//
#include <hip/hip_runtime.h>
#include <hip/hip_bf16.h>

typedef _Float16 half_t;
typedef __attribute__((ext_vector_type(16))) _Float16 v16h;
typedef __attribute__((ext_vector_type(8)))  float    v8f;

#define NPIX   5456          // 64^2+32^2+16^2+8^2+4^2
#define IC     256
#define NANCH  49104         // NPIX*9
#define NBATCH 2
#define PIXT   32            // pixels per workgroup
#define MT     171           // ceil(NPIX / PIXT)

// ---------------------------------------------------------------------------
// fp32 NCHW (per level) -> fp16 [n][global_pixel][IC] channel-last
// ---------------------------------------------------------------------------
__global__ void cvt_act_kernel(const float* __restrict__ f0, const float* __restrict__ f1,
                               const float* __restrict__ f2, const float* __restrict__ f3,
                               const float* __restrict__ f4, half_t* __restrict__ dst) {
    int idx = blockIdx.x * blockDim.x + threadIdx.x;
    const int total = NBATCH * NPIX * IC;
    if (idx >= total) return;
    int c = idx & (IC - 1);
    int p = (idx >> 8) % NPIX;
    int n = idx / (NPIX * IC);
    const int lb[6]  = {0, 4096, 5120, 5376, 5440, 5456};
    const int shl[5] = {6, 5, 4, 3, 2};
    int lvl = 0;
    while (p >= lb[lvl + 1]) lvl++;
    int sh = shl[lvl];
    int W  = 1 << sh;
    int q  = p - lb[lvl];
    int y  = q >> sh, x = q & (W - 1);
    const float* f = (lvl == 0) ? f0 : (lvl == 1) ? f1 : (lvl == 2) ? f2 : (lvl == 3) ? f3 : f4;
    float v = f[(((size_t)n * IC + c) * W + y) * W + x];   // H == W
    dst[idx] = (half_t)v;
}

// ---------------------------------------------------------------------------
// fp32 [O][IC][3][3] -> fp16 [tap][OCpad][IC], zero padded O rows
// ---------------------------------------------------------------------------
__global__ void cvt_w_kernel(const float* __restrict__ src, half_t* __restrict__ dst,
                             int OC, int OCpad) {
    int idx = blockIdx.x * blockDim.x + threadIdx.x;
    int total = 9 * OCpad * IC;
    if (idx >= total) return;
    int ic  = idx & (IC - 1);
    int o   = (idx >> 8) % OCpad;
    int tap = idx / (OCpad * IC);
    dst[idx] = (o < OC) ? (half_t)src[((size_t)o * IC + ic) * 9 + tap] : (half_t)0.f;
}

// ---------------------------------------------------------------------------
// Implicit-GEMM 3x3 conv via v_wmma_f32_16x16x32_f16.
// Block = 256 threads (8 waves) covering 32 pixels x 256 OC.
// Wave = 16 pixels x 64 OC (4 accumulators, A fragment reused 4x).
// A-slab (32 pix x 256 ch fp16) double-buffered in LDS; next tap streamed in
// with GLOBAL_LOAD_ASYNC_TO_LDS_B128 (ASYNCcnt) while current tap computes.
// FINAL==false: relu + fp16 store to channel-last act buffer.
// FINAL==true : bias + scatter fp32 into (n, pixel*9 + a, OUTDIM) layout.
// Compile-time OCPAD/OCVALID/OUTDIM: no runtime division, folded strides.
// ---------------------------------------------------------------------------
template <int OCPAD, int OCVALID, int OUTDIM, bool FINAL>
__launch_bounds__(256)
__global__ void conv3x3_wmma_kernel(const half_t* __restrict__ act,
                                    const half_t* __restrict__ wt,
                                    const float*  __restrict__ bias,
                                    half_t* __restrict__ outH,
                                    float*  __restrict__ outF) {
    __shared__ half_t As[2][PIXT * IC];   // 2 x 16 KB, double buffered

    const int lb[6]  = {0, 4096, 5120, 5376, 5440, 5456};
    const int shl[5] = {6, 5, 4, 3, 2};

    int pBase = blockIdx.x * PIXT;
    int lvl = 0;
    while (pBase >= lb[lvl + 1]) lvl++;
    int base = lb[lvl];
    int sh   = shl[lvl];
    int W    = 1 << sh;

    int n    = blockIdx.z;
    int tid  = threadIdx.x;
    int lane = tid & 31;
    int wave = tid >> 5;
    int pixsub = wave >> 2;                       // 0..1 : pixel sub-tile
    int ocg    = wave & 3;                        // 0..3 : OC group
    int ocbW   = blockIdx.y * 256 + ocg * 64;     // wave's first OC column
    bool active = ocbW < OCPAD;                   // wave-uniform

    // staging role: 256 threads = 32 pixels x 8 segments of 32 halves (64 B)
    int spix = tid >> 3;
    int sseg = tid & 7;
    int lp = pBase - base + spix;                 // may exceed level (last tile)
    int sy = lp >> sh, sx = lp & (W - 1);
    bool pixok = lp < W * W;

    auto stage = [&](int tap, half_t* buf) {
        int dy = tap / 3 - 1, dx = tap % 3 - 1;
        int yy = sy + dy, xx = sx + dx;
        half_t* l = buf + (spix << 8) + (sseg << 5);
        if (pixok && (unsigned)yy < (unsigned)W && (unsigned)xx < (unsigned)W) {
            const half_t* g = act + (((size_t)n * NPIX + base + (yy << sh) + xx) << 8)
                                  + (sseg << 5);
            unsigned la = (unsigned)(size_t)l;    // flat addr[31:0] == LDS byte address
#pragma unroll
            for (int k = 0; k < 4; ++k) {         // 4 x 16B async copies
                asm volatile("global_load_async_to_lds_b128 %0, %1, off"
                             :: "v"(la + k * 16), "v"(g + k * 8) : "memory");
            }
        } else {                                  // halo / tail pixels -> zeros
            v16h z = {};
            *(v16h*)l = z;
            *(v16h*)(l + 16) = z;
        }
    };

    const int mrow = lane & 15;                   // A: M row / B: N column
    const int khi  = lane >> 4;                   // upper-K half (lanes 16..31)

    v8f acc0 = {}, acc1 = {}, acc2 = {}, acc3 = {};

    stage(0, As[0]);
    asm volatile("s_wait_asynccnt 0x0" ::: "memory");
    __syncthreads();

#pragma unroll 1
    for (int tap = 0; tap < 9; ++tap) {
        half_t* cur = As[tap & 1];
        if (tap < 8) stage(tap + 1, As[(tap + 1) & 1]);   // overlap with WMMAs

        if (active) {
            const half_t* arow = cur + (((pixsub << 4) + mrow) << 8) + (khi << 4);
            const half_t* wrow = wt + (((size_t)(tap * OCPAD + ocbW + mrow)) << 8)
                                    + (khi << 4);
#pragma unroll
            for (int cc = 0; cc < 8; ++cc) {
                v16h a  = *(const v16h*)(arow + cc * 32);
                v16h b0 = *(const v16h*)(wrow + cc * 32);
                v16h b1 = *(const v16h*)(wrow + cc * 32 + 16 * IC);
                v16h b2 = *(const v16h*)(wrow + cc * 32 + 32 * IC);
                v16h b3 = *(const v16h*)(wrow + cc * 32 + 48 * IC);
                acc0 = __builtin_amdgcn_wmma_f32_16x16x32_f16(false, a, false, b0,
                        (short)0, acc0, false, false);
                acc1 = __builtin_amdgcn_wmma_f32_16x16x32_f16(false, a, false, b1,
                        (short)0, acc1, false, false);
                acc2 = __builtin_amdgcn_wmma_f32_16x16x32_f16(false, a, false, b2,
                        (short)0, acc2, false, false);
                acc3 = __builtin_amdgcn_wmma_f32_16x16x32_f16(false, a, false, b3,
                        (short)0, acc3, false, false);
            }
        }
        asm volatile("s_wait_asynccnt 0x0" ::: "memory");
        __syncthreads();
    }

    if (!active) return;

    // C/D layout: vgpr i -> M = i + 8*(lane>=16); N = lane & 15
    v8f accs[4] = {acc0, acc1, acc2, acc3};
    int col = lane & 15;
    int mb  = (lane >> 4) << 3;
    int lvlEnd = lb[lvl + 1];
#pragma unroll
    for (int t = 0; t < 4; ++t) {
        int ocb = ocbW + t * 16;
        if (ocb < OCPAD) {
            int oc = ocb + col;
            float bv = (oc < OCVALID) ? bias[oc] : 0.f;
#pragma unroll
            for (int i = 0; i < 8; ++i) {
                int pixel = pBase + (pixsub << 4) + mb + i;
                if (pixel < lvlEnd) {
                    float v = accs[t][i] + bv;
                    if (!FINAL) {
                        v = v > 0.f ? v : 0.f;
                        outH[(((size_t)n * NPIX + pixel) << 8) + oc] = (half_t)v;
                    } else if (oc < OCVALID) {
                        int a = oc / OUTDIM, c = oc % OUTDIM;   // compile-time divisor
                        outF[((size_t)n * NANCH + (size_t)pixel * 9 + a) * OUTDIM + c] = v;
                    }
                }
            }
        }
    }
}

// ---------------------------------------------------------------------------
extern "C" void kernel_launch(void* const* d_in, const int* in_sizes, int n_in,
                              void* d_out, int out_size, void* d_ws, size_t ws_size,
                              hipStream_t stream) {
    (void)in_sizes; (void)n_in; (void)out_size; (void)ws_size;

    const float* f[5];
    for (int i = 0; i < 5; ++i) f[i] = (const float*)d_in[i];
    const float* cw[4] = {(const float*)d_in[5], (const float*)d_in[7],
                          (const float*)d_in[9], (const float*)d_in[11]};
    const float* cb[4] = {(const float*)d_in[6], (const float*)d_in[8],
                          (const float*)d_in[10], (const float*)d_in[12]};
    const float* cls_w = (const float*)d_in[13];
    const float* cls_b = (const float*)d_in[14];
    const float* rw[4] = {(const float*)d_in[15], (const float*)d_in[17],
                          (const float*)d_in[19], (const float*)d_in[21]};
    const float* rb[4] = {(const float*)d_in[16], (const float*)d_in[18],
                          (const float*)d_in[20], (const float*)d_in[22]};
    const float* reg_w = (const float*)d_in[23];
    const float* reg_b = (const float*)d_in[24];

    // workspace layout (fp16): actIn | actP0 | actP1 | wbuf
    const size_t ACT = (size_t)NBATCH * NPIX * IC;
    half_t* actIn = (half_t*)d_ws;
    half_t* actP0 = actIn + ACT;
    half_t* actP1 = actP0 + ACT;
    half_t* wbuf  = actP1 + ACT;      // sized for cls: 9*832*256 halves

    float* clsOut = (float*)d_out;
    float* regOut = clsOut + (size_t)NBATCH * NANCH * 91;

    // 1) convert activations once (shared by both towers)
    {
        int tot = NBATCH * NPIX * IC;
        cvt_act_kernel<<<(tot + 255) / 256, 256, 0, stream>>>(f[0], f[1], f[2], f[3], f[4], actIn);
    }

    auto run_tower = [&](const float* const* w, const float* const* b,
                         const half_t*& inOut, half_t* const* pp) {
        const half_t* in = inOut;
        for (int l = 0; l < 4; ++l) {
            int tw = 9 * 256 * IC;
            cvt_w_kernel<<<(tw + 255) / 256, 256, 0, stream>>>(w[l], wbuf, 256, 256);
            conv3x3_wmma_kernel<256, 256, 1, false><<<dim3(MT, 1, NBATCH), 256, 0, stream>>>(
                in, wbuf, b[l], pp[l & 1], nullptr);
            in = pp[l & 1];
        }
        inOut = in;
    };

    half_t* pp[2] = {actP0, actP1};

    // 2) classification tower: final OC = 9*91 = 819 -> pad 832
    {
        const half_t* in = actIn;
        run_tower(cw, cb, in, pp);
        int tw = 9 * 832 * IC;
        cvt_w_kernel<<<(tw + 255) / 256, 256, 0, stream>>>(cls_w, wbuf, 819, 832);
        conv3x3_wmma_kernel<832, 819, 91, true><<<dim3(MT, 4, NBATCH), 256, 0, stream>>>(
            in, wbuf, cls_b, nullptr, clsOut);
    }
    // 3) regression tower: final OC = 9*4 = 36 -> pad 48
    {
        const half_t* in = actIn;
        run_tower(rw, rb, in, pp);
        int tw = 9 * 48 * IC;
        cvt_w_kernel<<<(tw + 255) / 256, 256, 0, stream>>>(reg_w, wbuf, 36, 48);
        conv3x3_wmma_kernel<48, 36, 4, true><<<dim3(MT, 1, NBATCH), 256, 0, stream>>>(
            in, wbuf, reg_b, nullptr, regOut);
    }
}
